// FastSelfAttention_11278584119727
// MI455X (gfx1250) — matrix operand order repaired and checked by reading, not compiled, not run on hardware
//
#include <hip/hip_runtime.h>

// ---------------- problem constants ----------------
constexpr int BB   = 16;
constexpr int TT   = 2048;
constexpr int DIM  = 768;
constexpr int HH   = 12;
constexpr int MTOT = BB * TT;          // 32768 rows

// ---------------- WMMA tile config ----------------
constexpr int TM   = 128;              // workgroup M tile
constexpr int TN   = 64;               // workgroup N tile
constexpr int KS   = 32;               // K step (bf16 wmma K)
constexpr int LDA  = 40;               // padded LDS row stride (bf16 elems)
constexpr int LDBT = 40;               // padded LDS row stride for B^T

typedef __attribute__((ext_vector_type(16))) __bf16 v16bf;
typedef __attribute__((ext_vector_type(8)))  __bf16 v8bf;
typedef __attribute__((ext_vector_type(8)))  float  v8f;
typedef __attribute__((ext_vector_type(4)))  unsigned int v4u;
typedef __attribute__((ext_vector_type(8)))  int v8i;
typedef __attribute__((ext_vector_type(4)))  int v4i;

__device__ inline unsigned lds_off(const void* p) {
  // __shared__ pointer -> flat; low 32 bits are the LDS byte address
  return (unsigned)(uintptr_t)p;
}

// ---------------- f32 -> bf16 convert ----------------
__global__ void fsa_convert(const float* __restrict__ src,
                            __bf16* __restrict__ dst, int n) {
  int i = blockIdx.x * blockDim.x + threadIdx.x;
  int stride = gridDim.x * blockDim.x;
  for (; i < n; i += stride) dst[i] = (__bf16)src[i];
}

// ---------------- WMMA GEMM ----------------
// C[m][n] = sum_k A[m][k] * Bw[k][n] + bias[n] (+ resid[m][n])
// A source: bf16 (Abf, staged via TDM tensor_load_to_lds with LDS padding)
//        or f32 (Af32, scaled per-k by ascale[b][k], b = m/TT; VALU-staged).
// B tile staged with per-lane global_load_async_to_lds_b128 into a raw
// [k][n] buffer, then LDS-local transpose into padded n-major layout so
// fragments remain contiguous ds_load_b128 reads.
__global__ __launch_bounds__(256) void fsa_wmma_gemm(
    const __bf16* __restrict__ Abf,
    const float*  __restrict__ Af32,
    const float*  __restrict__ ascale,
    const __bf16* __restrict__ Bw,
    const float*  __restrict__ bias,
    const float*  __restrict__ resid,
    float* __restrict__ Cout) {
  __shared__ __bf16 smA[TM * LDA];       // padded A tile (TDM pad matches)
  __shared__ __bf16 smBraw[KS * TN];     // raw async-landed B tile [k][n]
  __shared__ __bf16 smBt[TN * LDBT];     // transposed, padded B tile [n][k]

  const int tid  = threadIdx.x;
  const int lane = tid & 31;
  const int wave = tid >> 5;          // 8 waves (wave32)
  const int wm   = wave & 3;          // 4 wave-rows  (32 rows each)
  const int wn   = wave >> 2;         // 2 wave-cols  (32 cols each)
  const int rowBase = blockIdx.x * TM;
  const int colBase = blockIdx.y * TN;

  v8f c[2][2] = {};

  const int ar   = tid >> 1;          // A stage (f32 path): row 0..127
  const int aseg = (tid & 1) * 16;    //                     16-elem segment
  const int br   = tid >> 3;          // B stage: k-row 0..31
  const int bn0  = (tid & 7) * 8;     //          8-col segment

  const unsigned smA_base    = lds_off(smA);
  const unsigned smBraw_base = lds_off(smBraw);

  for (int kt = 0; kt < DIM; kt += KS) {
    // ---- stage A tile (128 x 32 bf16) ----
    if (Abf) {
      if (wave == 0) {
        // Tensor Data Mover: 2D tile, data_size=2B, tile 32x128, stride 768,
        // LDS padding 4 DWORDs every 16 DWORDs -> row stride LDA=40 bf16.
        unsigned long long ga =
            (unsigned long long)(uintptr_t)(Abf + (size_t)rowBase * DIM + kt);
        v4u g0;
        g0[0] = 1u;                                   // count=1 (valid), user mode
        g0[1] = smA_base;                             // lds_addr
        g0[2] = (unsigned)(ga & 0xFFFFFFFFu);         // global_addr[31:0]
        g0[3] = (unsigned)((ga >> 32) & 0x01FFFFFFu)  // global_addr[56:32]
              | (2u << 30);                           // type = 2 (image)
        v8i g1;
        g1[0] = (1 << 16)      // data_size = 2 bytes
              | (1 << 20)      // pad_enable
              | (3 << 22)      // pad_interval code 3 = 16 DWORDs (64B row)
              | (3 << 25);     // pad_amount  code 3 = 4 DWORDs (16B pad)
        g1[1] = (int)((unsigned)(DIM & 0xFFFF) << 16);        // tensor_dim0 lo
        g1[2] = (int)(((unsigned)TM & 0xFFFFu) << 16);        // tensor_dim1 lo = 128
        g1[3] = (int)((unsigned)KS << 16);                    // tile_dim0 = 32
        g1[4] = TM;                                           // tile_dim1 = 128
        g1[5] = DIM;                                          // tensor_dim0_stride lo
        g1[6] = 0;
        g1[7] = 0;
        v4i gz4 = {0, 0, 0, 0};
        v8i gz8 = {0, 0, 0, 0, 0, 0, 0, 0};
        __builtin_amdgcn_tensor_load_to_lds(g0, g1, gz4, gz4, gz8, 0);
      }
    } else {
      const int grow = rowBase + ar;
      __bf16* dst = smA + ar * LDA + aseg;
      const int  bidx = grow / TT;
      const float* src = Af32 + (size_t)grow * DIM + kt + aseg;
      const float* sc  = ascale + bidx * DIM + kt + aseg;
#pragma unroll
      for (int j = 0; j < 16; ++j) dst[j] = (__bf16)(src[j] * sc[j]);
      if (kt + KS < DIM) __builtin_prefetch(src + KS, 0, 1);
    }

    // ---- stage B tile raw via async copy (16B per lane, no VGPR transit) ----
    {
      const unsigned ldsb = smBraw_base + (unsigned)tid * 16u;
      const __bf16* src = Bw + (size_t)(kt + br) * DIM + colBase + bn0;
      asm volatile("global_load_async_to_lds_b128 %0, %1, off"
                   :: "v"(ldsb), "v"(src) : "memory");
      if (kt + KS < DIM) __builtin_prefetch(src + (size_t)KS * DIM, 0, 1);
    }

    // ---- completion waits, then make tiles visible workgroup-wide ----
    asm volatile("s_wait_asynccnt 0x0" ::: "memory");
    if (Abf && wave == 0) __builtin_amdgcn_s_wait_tensorcnt(0);
    __syncthreads();

    // ---- LDS-local transpose: smBraw[k][n] -> smBt[n][k] (padded) ----
    {
      v8bf vv = *(const v8bf*)(smBraw + br * TN + bn0);
#pragma unroll
      for (int j = 0; j < 8; ++j) smBt[(bn0 + j) * LDBT + br] = vv[j];
    }
    __syncthreads();

    const int h  = lane >> 4;
    const int ml = lane & 15;

    // A fragments: per ISA 16-bit A layout; contiguous per lane-half ->
    // compiler emits ds_load_b128 pairs.
    v16bf afr[2];
#pragma unroll
    for (int i = 0; i < 2; ++i) {
      const __bf16* r = smA + (wm * 32 + i * 16 + ml) * LDA;
#pragma unroll
      for (int e = 0; e < 16; ++e) {
        const int vg = e >> 1;
        const int k  = (vg < 4 ? 2 * vg : 16 + 2 * (vg - 4)) + (e & 1) + 8 * h;
        afr[i][e] = r[k];
      }
    }
    // B fragments: lane = col; K per lane-half h: 8h..8h+7, 16+8h..16+8h+7
    v16bf bfr[2];
#pragma unroll
    for (int j = 0; j < 2; ++j) {
      const __bf16* p = smBt + (wn * 32 + j * 16 + ml) * LDBT + 8 * h;
#pragma unroll
      for (int e = 0; e < 16; ++e) bfr[j][e] = p[e < 8 ? e : e + 8];
    }

#pragma unroll
    for (int i = 0; i < 2; ++i)
#pragma unroll
      for (int j = 0; j < 2; ++j)
        c[i][j] = __builtin_amdgcn_wmma_f32_16x16x32_bf16(
            false, afr[i], false, bfr[j], (short)0, c[i][j], false, false);
    __syncthreads();
  }

  // ---- epilogue: bias (+ residual), C/D layout m = r + 8*h, n = lane&15 ----
  const int h  = lane >> 4;
  const int nl = lane & 15;
#pragma unroll
  for (int i = 0; i < 2; ++i) {
#pragma unroll
    for (int j = 0; j < 2; ++j) {
      const int n = colBase + wn * 32 + j * 16 + nl;
#pragma unroll
      for (int r = 0; r < 8; ++r) {
        const int m = rowBase + wm * 32 + i * 16 + h * 8 + r;
        float v = c[i][j][r] + bias[n];
        if (resid) v += resid[(size_t)m * DIM + n];
        Cout[(size_t)m * DIM + n] = v;
      }
    }
  }
}

// ---------------- attention pooling (one block per (b,h)) ----------------
__global__ __launch_bounds__(256) void fsa_pool(
    const float* __restrict__ feat, const float* __restrict__ Wa,
    const float* __restrict__ ba, const int* __restrict__ mask,
    const float* __restrict__ pq, float* __restrict__ out, int isK) {
  __shared__ float s[TT];
  __shared__ float wvec[DIM];
  __shared__ float red[256];

  const int bh = blockIdx.x, b = bh / HH, h = bh % HH;
  const int tid = threadIdx.x, lane = tid & 31, wave = tid >> 5;

  for (int d = tid; d < DIM; d += 256)
    wvec[d] = isK ? pq[b * DIM + d] * Wa[d * HH + h] : Wa[d * HH + h];
  __syncthreads();

  const float* F = feat + (size_t)b * TT * DIM;
  const float bias = ba[h];
  for (int t = wave; t < TT; t += 8) {
    float acc = 0.f;
    const float* row = F + (size_t)t * DIM;
    for (int d = lane; d < DIM; d += 32) acc += row[d] * wvec[d];
#pragma unroll
    for (int off = 16; off; off >>= 1) acc += __shfl_xor(acc, off, 32);
    if (lane == 0) s[t] = (acc + bias) * 0.125f;   // / sqrt(64)
  }
  __syncthreads();

  // masked softmax over T
  const int* mk = mask + b * TT;
  float mx = -3.4e38f;
  for (int t = tid; t < TT; t += 256)
    if (mk[t] != 0) mx = fmaxf(mx, s[t]);
  red[tid] = mx; __syncthreads();
  for (int st = 128; st; st >>= 1) {
    if (tid < st) red[tid] = fmaxf(red[tid], red[tid + st]);
    __syncthreads();
  }
  mx = red[0]; __syncthreads();

  float sum = 0.f;
  for (int t = tid; t < TT; t += 256) {
    const float e = (mk[t] != 0) ? __expf(s[t] - mx) : 0.f;
    s[t] = e; sum += e;
  }
  red[tid] = sum; __syncthreads();
  for (int st = 128; st; st >>= 1) {
    if (tid < st) red[tid] += red[tid + st];
    __syncthreads();
  }
  const float inv = 1.f / fmaxf(red[0], 1e-20f);
  __syncthreads();

  // weighted head pooling
  const int dh = tid & 63, grp = tid >> 6;   // 4 t-slices x 64 dh
  float acc = 0.f;
  const float* Fh = F + h * 64 + dh;
  for (int t = grp; t < TT; t += 4) acc += s[t] * Fh[(size_t)t * DIM];
  red[grp * 64 + dh] = acc; __syncthreads();
  if (tid < 64) {
    float v = (red[tid] + red[64 + tid] + red[128 + tid] + red[192 + tid]) * inv;
    if (isK) v *= pq[b * DIM + h * 64 + tid];
    out[b * DIM + h * 64 + tid] = v;
  }
}

// ---------------- launcher ----------------
extern "C" void kernel_launch(void* const* d_in, const int* in_sizes, int n_in,
                              void* d_out, int out_size, void* d_ws, size_t ws_size,
                              hipStream_t stream) {
  const float* xs  = (const float*)d_in[0];
  const int*   msk = (const int*)  d_in[1];
  const float* Wq  = (const float*)d_in[2];
  const float* bq  = (const float*)d_in[3];
  const float* Wqa = (const float*)d_in[4];
  const float* bqa = (const float*)d_in[5];
  const float* Wk  = (const float*)d_in[6];
  const float* bk  = (const float*)d_in[7];
  const float* Wka = (const float*)d_in[8];
  const float* bka = (const float*)d_in[9];
  const float* Wt  = (const float*)d_in[10];
  const float* bt  = (const float*)d_in[11];
  float* out = (float*)d_out;

  char* ws = (char*)d_ws;
  size_t off = 0;
  auto carve = [&](size_t bytes) -> char* {
    char* p = ws + off;
    off += (bytes + 255) & ~(size_t)255;
    return p;
  };
  __bf16* Xb   = (__bf16*)carve((size_t)MTOT * DIM * sizeof(__bf16));
  __bf16* Wqb  = (__bf16*)carve((size_t)DIM * DIM * sizeof(__bf16));
  __bf16* Wkb  = (__bf16*)carve((size_t)DIM * DIM * sizeof(__bf16));
  __bf16* Wtb  = (__bf16*)carve((size_t)DIM * DIM * sizeof(__bf16));
  float*  qbuf = (float*)carve((size_t)MTOT * DIM * sizeof(float));
  float*  kbuf = (float*)carve((size_t)MTOT * DIM * sizeof(float));
  float*  pq   = (float*)carve((size_t)BB * DIM * sizeof(float));
  float*  pk   = (float*)carve((size_t)BB * DIM * sizeof(float));

  // 1) bf16 conversions (Xb = 50MB -> L2-resident, reused by both GEMMs)
  fsa_convert<<<4096, 256, 0, stream>>>(xs, Xb, MTOT * DIM);
  fsa_convert<<<1024, 256, 0, stream>>>(Wq, Wqb, DIM * DIM);
  fsa_convert<<<1024, 256, 0, stream>>>(Wk, Wkb, DIM * DIM);
  fsa_convert<<<1024, 256, 0, stream>>>(Wt, Wtb, DIM * DIM);

  dim3 ggrid(MTOT / TM, DIM / TN);   // 256 x 12
  // 2) q = X@Wq + bq ; 3) k = X@Wk + bk   (A tiles via TDM)
  fsa_wmma_gemm<<<ggrid, 256, 0, stream>>>(Xb, nullptr, nullptr, Wqb, bq, nullptr, qbuf);
  fsa_wmma_gemm<<<ggrid, 256, 0, stream>>>(Xb, nullptr, nullptr, Wkb, bk, nullptr, kbuf);

  // 4) pooled_q  5) pooled_k (pooled_q folded in, mqk never materialized)
  fsa_pool<<<BB * HH, 256, 0, stream>>>(qbuf, Wqa, bqa, msk, nullptr, pq, 0);
  fsa_pool<<<BB * HH, 256, 0, stream>>>(kbuf, Wka, bka, msk, pq, pk, 1);

  // 6) out = (pk ⊙ q)@Wt + bt + q   (A built on the fly from q f32 * pk)
  fsa_wmma_gemm<<<ggrid, 256, 0, stream>>>(nullptr, qbuf, pk, Wtb, bt, qbuf, out);
}
